// Attention_74131135529194
// MI455X (gfx1250) — matrix operand order repaired and checked by reading, not compiled, not run on hardware
//
#include <hip/hip_runtime.h>
#include <math.h>

// ---------------------------------------------------------------------------
// Gated attention for MI455X (gfx1250, wave32, WMMA + TDM).
//  - all matmuls: v_wmma_f32_16x16x32_bf16 (fp32 accum)
//  - all B operands pre-transposed so fragment loads are contiguous b128s
//  - src_mask tile DMA'd into LDS by the Tensor Data Mover, overlapped with
//    the Q.K^T WMMA phase; softmax/blend runs out of LDS.
// ---------------------------------------------------------------------------

typedef __attribute__((ext_vector_type(16))) __bf16 v16bf;
typedef __attribute__((ext_vector_type(8)))  __bf16 v8bf;
typedef __attribute__((ext_vector_type(8)))  float  v8f;
typedef __attribute__((ext_vector_type(8)))  float  v8fl;
typedef __attribute__((ext_vector_type(4)))  unsigned int u32x4;
typedef __attribute__((ext_vector_type(8)))  int i32x8;
typedef __attribute__((ext_vector_type(4)))  int i32x4;

#define BB 4
#define LL 1024
#define DD 1024
#define HH 8
#define DHD 128
#define QK_SCALE 0.03125f   // 1024^-0.5 (reference scales by full dim_out)

#if defined(__has_builtin)
#  if __has_builtin(__builtin_amdgcn_tensor_load_to_lds)
#    define HAVE_TDM 1
#  endif
#endif
#ifndef HAVE_TDM
#  define HAVE_TDM 0
#endif

// ---- WMMA fragment loaders (contiguous k = 16*(lane/16) + s in both A and B, wave32) ---------

// A tile 16x32 (MxK), row-major bf16: two 16B vector loads per lane.
static __device__ __forceinline__ v16bf load_a_bf16(const __bf16* base, int ld,
                                                    int m0, int k0, int lane) {
  int r = lane & 15;
  int koff = (lane >> 4) * 16;
  const v8bf* p = (const v8bf*)(base + (size_t)(m0 + r) * ld + k0 + koff);
  v8bf lo = p[0];
  v8bf hi = p[1];   // +8 elements
  return __builtin_shufflevector(lo, hi, 0, 1, 2, 3, 4, 5, 6, 7,
                                 8, 9, 10, 11, 12, 13, 14, 15);
}

// A tile from fp32 source (convert on load); two 32B vector loads per lane.
static __device__ __forceinline__ v16bf load_a_f32(const float* base, int ld,
                                                   int m0, int k0, int lane) {
  int r = lane & 15;
  int koff = (lane >> 4) * 16;
  const v8fl* p = (const v8fl*)(base + (size_t)(m0 + r) * ld + k0 + koff);
  v8fl lo = p[0];
  v8fl hi = p[1];
  v16bf a;
#pragma unroll
  for (int i = 0; i < 8; ++i) { a[i] = (__bf16)lo[i]; a[i + 8] = (__bf16)hi[i]; }
  return a;
}

// B tile 32x16 (KxN) where the source is stored N-major (i.e. B^T, N x K):
// element (k,n) = rows[n0+n][k0+k] -> one contiguous 32B load per lane.
static __device__ __forceinline__ v16bf load_b_rowT(const __bf16* rows, int ld,
                                                    int k0, int n0, int lane) {
  int n = lane & 15;
  int kb = k0 + (lane >> 4) * 16;
  return *(const v16bf*)(rows + (size_t)(n0 + n) * ld + kb);
}

// ---- kernels --------------------------------------------------------------

__global__ void cast_f32_to_bf16(const float* __restrict__ in,
                                 __bf16* __restrict__ out, int n) {
  for (int i = blockIdx.x * blockDim.x + threadIdx.x; i < n;
       i += gridDim.x * blockDim.x)
    out[i] = (__bf16)in[i];
}

// out(N x K, bf16) = transpose(in(K x N, f32)); 32x32 LDS tile, 256 threads.
__global__ __launch_bounds__(256)
void cast_transpose_f32_bf16(const float* __restrict__ in,
                             __bf16* __restrict__ out, int R, int C) {
  __shared__ float tile[32][33];
  int c0 = blockIdx.x * 32, r0 = blockIdx.y * 32;
  int tx = threadIdx.x & 31, ty = threadIdx.x >> 5;
  for (int rr = ty; rr < 32; rr += 8)
    tile[rr][tx] = in[(size_t)(r0 + rr) * C + c0 + tx];
  __syncthreads();
  for (int rr = ty; rr < 32; rr += 8)
    out[(size_t)(c0 + rr) * R + r0 + tx] = (__bf16)tile[tx][rr];
}

// C(bf16, MxN) = A(bf16, MxK) * B with B given transposed (BT: N x K).
__global__ __launch_bounds__(256)
void gemm_bf16(const __bf16* __restrict__ A, const __bf16* __restrict__ BT,
               __bf16* __restrict__ C, int M, int N, int K) {
  int lane = threadIdx.x & 31;
  int wave = threadIdx.x >> 5;
  int tile = blockIdx.x * 8 + wave;
  int tn = N >> 4;
  int m0 = (tile / tn) << 4;
  int n0 = (tile % tn) << 4;
  if (m0 >= M) return;
  v8f acc = {};
  for (int k0 = 0; k0 < K; k0 += 32) {
    v16bf a = load_a_bf16(A, K, m0, k0, lane);
    v16bf b = load_b_rowT(BT, K, k0, n0, lane);
    acc = __builtin_amdgcn_wmma_f32_16x16x32_bf16(false, a, false, b,
                                                  (short)0, acc, false, false);
  }
  int n = lane & 15, mh = (lane >> 4) * 8;
#pragma unroll
  for (int r = 0; r < 8; ++r)
    C[(size_t)(m0 + mh + r) * N + n0 + n] = (__bf16)acc[r];
}

// Same GEMM but stores the result transposed per batch: CT[b, n, l] with
// m = b*LL + l. Used to produce V^T so attn@V B-fragments are contiguous.
__global__ __launch_bounds__(256)
void gemm_bf16_storeT(const __bf16* __restrict__ A, const __bf16* __restrict__ BT,
                      __bf16* __restrict__ CT, int M, int N, int K) {
  int lane = threadIdx.x & 31;
  int wave = threadIdx.x >> 5;
  int tile = blockIdx.x * 8 + wave;
  int tn = N >> 4;
  int m0 = (tile / tn) << 4;
  int n0 = (tile % tn) << 4;
  if (m0 >= M) return;
  v8f acc = {};
  for (int k0 = 0; k0 < K; k0 += 32) {
    v16bf a = load_a_bf16(A, K, m0, k0, lane);
    v16bf b = load_b_rowT(BT, K, k0, n0, lane);
    acc = __builtin_amdgcn_wmma_f32_16x16x32_bf16(false, a, false, b,
                                                  (short)0, acc, false, false);
  }
  int n = lane & 15, mh = (lane >> 4) * 8;
#pragma unroll
  for (int r = 0; r < 8; ++r) {
    int m = m0 + mh + r;
    CT[((size_t)(m >> 10) * N + n0 + n) * LL + (m & 1023)] = (__bf16)acc[r];
  }
}

// C(f32) = A(bf16) * B (BT given) + bias. Final projection.
__global__ __launch_bounds__(256)
void gemm_bf16_bias_f32(const __bf16* __restrict__ A, const __bf16* __restrict__ BT,
                        const float* __restrict__ bias, float* __restrict__ C,
                        int M, int N, int K) {
  int lane = threadIdx.x & 31;
  int wave = threadIdx.x >> 5;
  int tile = blockIdx.x * 8 + wave;
  int tn = N >> 4;
  int m0 = (tile / tn) << 4;
  int n0 = (tile % tn) << 4;
  if (m0 >= M) return;
  v8f acc = {};
  for (int k0 = 0; k0 < K; k0 += 32) {
    v16bf a = load_a_bf16(A, K, m0, k0, lane);
    v16bf b = load_b_rowT(BT, K, k0, n0, lane);
    acc = __builtin_amdgcn_wmma_f32_16x16x32_bf16(false, a, false, b,
                                                  (short)0, acc, false, false);
  }
  int n = lane & 15, mh = (lane >> 4) * 8;
  float bv = bias[n0 + n];
#pragma unroll
  for (int r = 0; r < 8; ++r)
    C[(size_t)(m0 + mh + r) * N + n0 + n] = acc[r] + bv;
}

// Fused: TDM prefetch of 16x1024 src_mask tile into LDS, Q.K^T via WMMA into
// LDS, row softmax of both, sigmoid-gated blend, write attn.
// 512 threads = 16 waves; WG owns a 16-row block of one (b,h).
// The reference's renormalize is the identity ((1-g)*1 + g*1 == 1) -> dropped.
__global__ __launch_bounds__(512)
void attn_score_softmax(const __bf16* __restrict__ Qb, const __bf16* __restrict__ Kb,
                        const float* __restrict__ src_mask,
                        const float* __restrict__ gating,
                        float* __restrict__ attn) {
  __shared__ float s_sc[16 * LL];    // 64 KB score block
  __shared__ float s_mask[16 * LL];  // 64 KB mask block (TDM destination)
  int bh = blockIdx.y;               // 0..31
  int b = bh >> 3, h = bh & 7;
  int l0 = blockIdx.x << 4;
  int lane = threadIdx.x & 31;
  int wave = threadIdx.x >> 5;       // 0..15

  const __bf16* Qh = Qb + (size_t)b * LL * DD + h * DHD;
  const __bf16* Kh = Kb + (size_t)b * LL * DD + h * DHD;

  // --- kick off the src_mask tile DMA (overlaps with the WMMA phase) ---
#if HAVE_TDM
  if (wave == 0) {
    unsigned long long ga =
        (unsigned long long)(uintptr_t)(src_mask + ((size_t)bh * LL + l0) * LL);
    unsigned ldsoff = (unsigned)((const char*)s_mask - (const char*)s_sc);
    u32x4 g0 = {};
    g0[0] = 1u;                                   // count=1 valid descriptor
    g0[1] = ldsoff;                               // lds_addr
    g0[2] = (unsigned)(ga & 0xFFFFFFFFu);         // global_addr[31:0]
    g0[3] = (unsigned)((ga >> 32) & 0x01FFFFFFu)  // global_addr[56:32]
            | (2u << 30);                         // type=2 (image)
    i32x8 g1 = {};
    g1[0] = (int)(2u << 16);                      // data_size=4B, no multicast
    g1[1] = (int)((unsigned)LL << 16);            // tensor_dim0 = 1024
    g1[2] = (int)((unsigned)LL << 16);            // tensor_dim1 = 1024
    g1[3] = (int)((unsigned)LL << 16);            // tile_dim0   = 1024
    g1[4] = 16;                                   // tile_dim1   = 16
    g1[5] = LL;                                   // tensor_dim0_stride = 1024
    i32x4 z4 = {};
#if defined(__clang_major__) && __clang_major__ >= 23
    i32x8 z8 = {};
    __builtin_amdgcn_tensor_load_to_lds(g0, g1, z4, z4, z8, 0);
#else
    __builtin_amdgcn_tensor_load_to_lds(g0, g1, z4, z4, 0);
#endif
  }
#else
  {
    const float* mb = src_mask + ((size_t)bh * LL + l0) * LL;
    for (int i = threadIdx.x; i < 16 * LL; i += 512) s_mask[i] = mb[i];
  }
#endif

  // --- score block via WMMA: each wave does 4 column tiles ---
#pragma unroll
  for (int tt = 0; tt < 4; ++tt) {
    int t0 = (wave * 4 + tt) << 4;
    v8f acc = {};
#pragma unroll
    for (int d0 = 0; d0 < DHD; d0 += 32) {
      v16bf a = load_a_bf16(Qh, DD, l0, d0, lane);
      v16bf kb = load_b_rowT(Kh, DD, d0, t0, lane);
      acc = __builtin_amdgcn_wmma_f32_16x16x32_bf16(false, a, false, kb,
                                                    (short)0, acc, false, false);
    }
    int n = lane & 15, mh = (lane >> 4) * 8;
#pragma unroll
    for (int r = 0; r < 8; ++r)
      s_sc[(mh + r) * LL + t0 + n] = acc[r] * QK_SCALE;
  }

#if HAVE_TDM
  if (wave == 0) __builtin_amdgcn_s_wait_tensorcnt(0);  // DMA done before barrier
#endif
  __syncthreads();

  // --- wave r owns row r: softmax(score), softmax(mask), blend ---
  int row = wave;
  int l = l0 + row;
  float* srow = s_sc + row * LL;
  float* mrow = s_mask + row * LL;

  float mx = -3.0e38f;
  for (int i = lane; i < LL; i += 32) mx = fmaxf(mx, srow[i]);
#pragma unroll
  for (int o = 16; o > 0; o >>= 1) mx = fmaxf(mx, __shfl_xor(mx, o));
  float sum = 0.f;
  for (int i = lane; i < LL; i += 32) {
    float e = __expf(srow[i] - mx);
    srow[i] = e;
    sum += e;
  }
#pragma unroll
  for (int o = 16; o > 0; o >>= 1) sum += __shfl_xor(sum, o);
  float sinv = 1.f / sum;

  float mmx = -3.0e38f;
  for (int i = lane; i < LL; i += 32) mmx = fmaxf(mmx, mrow[i]);
#pragma unroll
  for (int o = 16; o > 0; o >>= 1) mmx = fmaxf(mmx, __shfl_xor(mmx, o));
  float msum = 0.f;
  for (int i = lane; i < LL; i += 32) {
    float e = __expf(mrow[i] - mmx);
    mrow[i] = e;
    msum += e;
  }
#pragma unroll
  for (int o = 16; o > 0; o >>= 1) msum += __shfl_xor(msum, o);
  float minv = 1.f / msum;

  float g = 1.f / (1.f + __expf(-gating[h]));
  float* arow = attn + ((size_t)bh * LL + l) * LL;
  for (int i = lane; i < LL; i += 32)
    arow[i] = (1.f - g) * srow[i] * sinv + g * mrow[i] * minv;
}

// out_heads(bf16) = attn(f32) * V, with V given transposed (VT: B x D x L).
__global__ __launch_bounds__(256)
void attn_v(const float* __restrict__ attn, const __bf16* __restrict__ VT,
            __bf16* __restrict__ OH) {
  int bh = blockIdx.y;
  int b = bh >> 3, h = bh & 7;
  int lane = threadIdx.x & 31;
  int wave = threadIdx.x >> 5;
  int tile = blockIdx.x * 8 + wave;   // 0..511
  int l0 = (tile >> 3) << 4;          // 64 row tiles
  int n0 = (tile & 7) << 4;           // 8 col tiles (128-wide head)
  const float* Ah = attn + (size_t)bh * LL * LL;
  const __bf16* Vh = VT + ((size_t)b * DD + h * DHD) * LL;
  v8f acc = {};
  for (int k0 = 0; k0 < LL; k0 += 32) {
    v16bf a = load_a_f32(Ah, LL, l0, k0, lane);
    v16bf vb = load_b_rowT(Vh, LL, k0, n0, lane);
    acc = __builtin_amdgcn_wmma_f32_16x16x32_bf16(false, a, false, vb,
                                                  (short)0, acc, false, false);
  }
  int n = lane & 15, mh = (lane >> 4) * 8;
#pragma unroll
  for (int r = 0; r < 8; ++r)
    OH[((size_t)b * LL + l0 + mh + r) * DD + h * DHD + n0 + n] = (__bf16)acc[r];
}

// ---------------------------------------------------------------------------

extern "C" void kernel_launch(void* const* d_in, const int* in_sizes, int n_in,
                              void* d_out, int out_size, void* d_ws, size_t ws_size,
                              hipStream_t stream) {
  (void)in_sizes; (void)n_in; (void)out_size; (void)ws_size;
  const float* x      = (const float*)d_in[0];
  const float* mask   = (const float*)d_in[1];
  const float* Wq     = (const float*)d_in[2];
  const float* Wk     = (const float*)d_in[3];
  const float* Wv     = (const float*)d_in[4];
  const float* Wp     = (const float*)d_in[5];
  const float* bp     = (const float*)d_in[6];
  const float* gating = (const float*)d_in[7];

  float* out  = (float*)d_out;                     // (B,L,D)
  float* attn = out + (size_t)BB * LL * DD;        // (B*H,L,L)

  // workspace layout (48 MB total, fully rewritten each call)
  __bf16* xb   = (__bf16*)d_ws;                    // (B*L, D)
  __bf16* wqT  = xb   + (size_t)BB * LL * DD;      // (D, D) transposed
  __bf16* wkT  = wqT  + (size_t)DD * DD;
  __bf16* wvT  = wkT  + (size_t)DD * DD;
  __bf16* wpT  = wvT  + (size_t)DD * DD;
  __bf16* Qb   = wpT  + (size_t)DD * DD;           // (B*L, D)
  __bf16* Kb   = Qb   + (size_t)BB * LL * DD;      // (B*L, D)
  __bf16* VT   = Kb   + (size_t)BB * LL * DD;      // (B, D, L)
  __bf16* OH   = VT   + (size_t)BB * LL * DD;      // (B*L, D)

  cast_f32_to_bf16<<<1024, 256, 0, stream>>>(x, xb, BB * LL * DD);
  dim3 gt(DD / 32, DD / 32);
  cast_transpose_f32_bf16<<<gt, 256, 0, stream>>>(Wq, wqT, DD, DD);
  cast_transpose_f32_bf16<<<gt, 256, 0, stream>>>(Wk, wkT, DD, DD);
  cast_transpose_f32_bf16<<<gt, 256, 0, stream>>>(Wv, wvT, DD, DD);
  cast_transpose_f32_bf16<<<gt, 256, 0, stream>>>(Wp, wpT, DD, DD);

  const int M = BB * LL;                            // 4096
  const int gemm_blocks = (M / 16) * (DD / 16) / 8; // 2048 WGs of 8 waves
  gemm_bf16<<<gemm_blocks, 256, 0, stream>>>(xb, wqT, Qb, M, DD, DD);
  gemm_bf16<<<gemm_blocks, 256, 0, stream>>>(xb, wkT, Kb, M, DD, DD);
  gemm_bf16_storeT<<<gemm_blocks, 256, 0, stream>>>(xb, wvT, VT, M, DD, DD);

  dim3 gs(LL / 16, BB * HH);                        // (64, 32)
  attn_score_softmax<<<gs, 512, 0, stream>>>(Qb, Kb, mask, gating, attn);

  dim3 gv(64, BB * HH);                             // 512 tiles / 8 waves per WG
  attn_v<<<gv, 256, 0, stream>>>(attn, VT, OH);

  gemm_bf16_bias_f32<<<gemm_blocks, 256, 0, stream>>>(OH, wpT, bp, out, M, DD, DD);
}
